// SkipTransformer_87101936763055
// MI455X (gfx1250) — compile-verified, matrix operand order checked by reading, not run on hardware
//
#include <hip/hip_runtime.h>
#include <hip/hip_bf16.h>
#include <math.h>
#include <stdint.h>

// ---------------- problem constants ----------------
#define B_    2
#define C_    128
#define N_    4096
#define K_    16
#define DIM_  256
#define PH_   64
#define AH_   1024
#define NK_   (N_ * K_)          // 65536
#define EPS_  1e-5f

typedef __bf16 bf16;
typedef __bf16 v16bf __attribute__((ext_vector_type(16)));
typedef __bf16 v8bf  __attribute__((ext_vector_type(8)));
typedef float  v8f   __attribute__((ext_vector_type(8)));

// bf16 weight arena offsets (elements)
enum {
  OW_MLPV_W1 = 0,        // 128x256
  OW_MLPV_W2 = 32768,    // 128x128
  OW_MLPV_WS = 49152,    // 128x256
  OW_WK      = 81920,    // 256x128
  OW_WQ      = 114688,   // 256x128
  OW_WV      = 147456,   // 256x128
  OW_POSW2   = 180224,   // 256x64
  OW_ATTW1   = 196608,   // 1024x256
  OW_ATTW2   = 458752,   // 256x1024
  OW_WE      = 720896,   // 128x256
  W_TOTAL    = 753664
};

// ---------------- WMMA helpers ----------------
// CDNA5 16-bit A/B fragment (16x32 MxK): lanes 0-15 -> row r, K = k0..k0+7 and
// k0+16..k0+23 ; lanes 16-31 -> row r, K = k0+8..k0+15 and k0+24..k0+31.
// With row-contiguous bf16 data this is exactly two 16-byte loads per lane.
__device__ __forceinline__ v16bf load_frag(const bf16* base, int stride, int row0, int k0) {
  const int lane = threadIdx.x & 31;
  const int r = lane & 15, hi = lane >> 4;
  const bf16* p = base + (size_t)(row0 + r) * stride + (k0 + hi * 8);
  union { v16bf v; v8bf h[2]; } u;
  u.h[0] = *reinterpret_cast<const v8bf*>(p);
  u.h[1] = *reinterpret_cast<const v8bf*>(p + 16);
  return u.v;
}

__device__ __forceinline__ v8f wmma_bf16(v16bf a, v16bf b, v8f c) {
  return __builtin_amdgcn_wmma_f32_16x16x32_bf16(false, a, false, b, (short)0, c, false, false);
}

__device__ __forceinline__ float red16_max(float x) {
  x = fmaxf(x, __shfl_xor(x, 1));
  x = fmaxf(x, __shfl_xor(x, 2));
  x = fmaxf(x, __shfl_xor(x, 4));
  x = fmaxf(x, __shfl_xor(x, 8));
  return x;
}
__device__ __forceinline__ float red16_sum(float x) {
  x += __shfl_xor(x, 1);
  x += __shfl_xor(x, 2);
  x += __shfl_xor(x, 4);
  x += __shfl_xor(x, 8);
  return x;
}

// ---------------- async global->LDS staging (CDNA5 async-tensor path) ----------------
// Copies `bytes` (multiple of 256*16) from global to LDS with per-lane
// GLOBAL_LOAD_ASYNC_TO_LDS_B128, retired with s_wait_asynccnt. Caller must
// __syncthreads() before other waves consume the panel.
__device__ __forceinline__ void async_stage(const bf16* gsrc, bf16* lds_dst, int bytes) {
  uint64_t g = (uint64_t)(size_t)gsrc;
  uint32_t l = (uint32_t)(size_t)lds_dst;     // flat LDS addr low bits == LDS offset
  for (int o = threadIdx.x * 16; o < bytes; o += 256 * 16) {
    asm volatile("global_load_async_to_lds_b128 %0, %1, off"
                 :: "v"(l + (uint32_t)o), "v"(g + (uint64_t)o) : "memory");
  }
  asm volatile("s_wait_asynccnt 0x0" ::: "memory");
}

// ---------------- small utility kernels ----------------
__global__ void cvt_f32_bf16_kernel(const float* __restrict__ src, bf16* __restrict__ dst, int n) {
  int i = blockIdx.x * blockDim.x + threadIdx.x;
  if (i < n) dst[i] = (bf16)src[i];
}

// xT[b][n][c] bf16, c: 0..127 = key, 128..255 = query
__global__ void pack_x_kernel(const float* __restrict__ key, const float* __restrict__ query,
                              bf16* __restrict__ xT) {
  int i = blockIdx.x * blockDim.x + threadIdx.x;   // B*N*256
  int c = i & 255;
  int n = (i >> 8) & (N_ - 1);
  int b = i >> 20;
  float v = (c < C_) ? key[((size_t)b * C_ + c) * N_ + n]
                     : query[((size_t)b * C_ + (c - C_)) * N_ + n];
  xT[i] = (bf16)v;
}

// ---------------- value MLP ----------------
// h1 = relu(W1 @ x + b1)  -> h1T bf16 (B,N,128)
__global__ __launch_bounds__(256)
void gemm_h1_kernel(const bf16* __restrict__ wbf, const bf16* __restrict__ xT,
                    const float* __restrict__ b1, bf16* __restrict__ h1T) {
  int gw = (blockIdx.x * blockDim.x + threadIdx.x) >> 5;
  int b = gw >> 11, t = gw & 2047;                 // 8*256 tiles per batch
  int m0 = (t & 7) << 4, c0 = (t >> 3) << 4;
  const bf16* W = wbf + OW_MLPV_W1;
  const bf16* X = xT + (size_t)b * N_ * 256;
  v8f acc = {};
  for (int k = 0; k < 256; k += 32)
    acc = wmma_bf16(load_frag(W, 256, m0, k), load_frag(X, 256, c0, k), acc);
  int lane = threadIdx.x & 31, r = lane & 15, hi = lane >> 4;
  bf16* out = h1T + ((size_t)b * N_ + (c0 + r)) * C_;
#pragma unroll
  for (int i = 0; i < 8; ++i) {
    int row = m0 + i + 8 * hi;
    out[row] = (bf16)fmaxf(acc[i] + b1[row], 0.f);
  }
}

// value = W2 @ h1 + b2 + Ws @ x + bs   -> value f32 (B,C,N) and valT bf16 (B,N,C)
__global__ __launch_bounds__(256)
void gemm_value_kernel(const bf16* __restrict__ wbf, const bf16* __restrict__ xT,
                       const bf16* __restrict__ h1T, const float* __restrict__ b2,
                       const float* __restrict__ bs, float* __restrict__ value,
                       bf16* __restrict__ valT) {
  int gw = (blockIdx.x * blockDim.x + threadIdx.x) >> 5;
  int b = gw >> 11, t = gw & 2047;
  int m0 = (t & 7) << 4, c0 = (t >> 3) << 4;
  const bf16* X = xT + (size_t)b * N_ * 256;
  const bf16* H = h1T + (size_t)b * N_ * C_;
  v8f acc = {};
  for (int k = 0; k < 128; k += 32)
    acc = wmma_bf16(load_frag(wbf + OW_MLPV_W2, 128, m0, k), load_frag(H, 128, c0, k), acc);
  for (int k = 0; k < 256; k += 32)
    acc = wmma_bf16(load_frag(wbf + OW_MLPV_WS, 256, m0, k), load_frag(X, 256, c0, k), acc);
  int lane = threadIdx.x & 31, r = lane & 15, hi = lane >> 4;
  int col = c0 + r;
#pragma unroll
  for (int i = 0; i < 8; ++i) {
    int row = m0 + i + 8 * hi;
    float v = acc[i] + b2[row] + bs[row];
    value[((size_t)b * C_ + row) * N_ + col] = v;
    valT[((size_t)b * N_ + col) * C_ + row] = (bf16)v;
  }
}

// ---------------- k/q/v projections (M=256, kdim=128) -> outT f32 (B,N,256) ----------------
__global__ __launch_bounds__(256)
void gemm_proj_kernel(const bf16* __restrict__ W, const bf16* __restrict__ Xt, int xstride,
                      const float* __restrict__ bias, float* __restrict__ outT) {
  int gw = (blockIdx.x * blockDim.x + threadIdx.x) >> 5;
  int b = gw >> 12, t = gw & 4095;                 // 16*256 tiles per batch
  int m0 = (t & 15) << 4, c0 = (t >> 4) << 4;
  const bf16* X = Xt + (size_t)b * N_ * xstride;
  v8f acc = {};
  for (int k = 0; k < 128; k += 32)
    acc = wmma_bf16(load_frag(W, 128, m0, k), load_frag(X, xstride, c0, k), acc);
  int lane = threadIdx.x & 31, r = lane & 15, hi = lane >> 4;
  float* out = outT + ((size_t)b * N_ + (c0 + r)) * DIM_;
#pragma unroll
  for (int i = 0; i < 8; ++i) {
    int row = m0 + i + 8 * hi;
    out[row] = acc[i] + bias[row];
  }
}

// ---------------- KNN (top-16 smallest squared distance, self included) ----------------
__global__ __launch_bounds__(256)
void knn_kernel(const float* __restrict__ pos, int* __restrict__ idx) {
  __shared__ float px[N_], py[N_], pz[N_];
  int b = blockIdx.x >> 4;
  int n = ((blockIdx.x & 15) << 8) + threadIdx.x;
  const float* P = pos + (size_t)b * 3 * N_;
  for (int j = threadIdx.x; j < N_; j += blockDim.x) {
    px[j] = P[j]; py[j] = P[N_ + j]; pz[j] = P[2 * N_ + j];
  }
  __syncthreads();
  float qx = px[n], qy = py[n], qz = pz[n];
  float bd[K_]; int bi[K_];
#pragma unroll
  for (int k = 0; k < K_; ++k) { bd[k] = 3.4e38f; bi[k] = 0; }
  for (int j = 0; j < N_; ++j) {
    float dx = qx - px[j], dy = qy - py[j], dz = qz - pz[j];
    float d = dx * dx + dy * dy + dz * dz;
    if (d < bd[K_ - 1]) {
      int p = K_ - 1;
      while (p > 0 && bd[p - 1] > d) { bd[p] = bd[p - 1]; bi[p] = bi[p - 1]; --p; }
      bd[p] = d; bi[p] = j;
    }
  }
  int* o = idx + ((size_t)b * N_ + n) * K_;
#pragma unroll
  for (int k = 0; k < K_; ++k) o[k] = bi[k];
}

// ---------------- positional MLP hidden: p1t f32 (B*NK, 64) ----------------
__global__ __launch_bounds__(256)
void pos_hidden_kernel(const float* __restrict__ pos, const int* __restrict__ idx,
                       const float* __restrict__ w1, const float* __restrict__ b1,
                       float* __restrict__ p1t) {
  __shared__ float sw[PH_ * 3], sb[PH_];
  int tid = threadIdx.x;
  if (tid < PH_ * 3) sw[tid] = w1[tid];
  if (tid < PH_) sb[tid] = b1[tid];
  __syncthreads();
  int gcol = blockIdx.x * 256 + tid;               // b*NK + col
  int b = gcol >> 16;
  int col = gcol & (NK_ - 1);
  int n = col >> 4;
  int j = idx[gcol];
  const float* P = pos + (size_t)b * 3 * N_;
  float dx = P[n] - P[j], dy = P[N_ + n] - P[N_ + j], dz = P[2 * N_ + n] - P[2 * N_ + j];
  float* out = p1t + (size_t)gcol * PH_;
#pragma unroll
  for (int c = 0; c < PH_; ++c)
    out[c] = sw[c * 3] * dx + sw[c * 3 + 1] * dy + sw[c * 3 + 2] * dz + sb[c];
}

// deterministic per-channel sum/sumsq over p1t (one block per channel)
__global__ __launch_bounds__(256)
void pos_stats_kernel(const float* __restrict__ p1t, float* __restrict__ sum,
                      float* __restrict__ sq) {
  __shared__ float s1[256], s2[256];
  int c = blockIdx.x;
  float a = 0.f, q = 0.f;
  for (int i = threadIdx.x; i < B_ * NK_; i += 256) {
    float v = p1t[(size_t)i * PH_ + c];
    a += v; q += v * v;
  }
  s1[threadIdx.x] = a; s2[threadIdx.x] = q;
  __syncthreads();
  for (int st = 128; st > 0; st >>= 1) {
    if (threadIdx.x < st) {
      s1[threadIdx.x] += s1[threadIdx.x + st];
      s2[threadIdx.x] += s2[threadIdx.x + st];
    }
    __syncthreads();
  }
  if (threadIdx.x == 0) { sum[c] = s1[0]; sq[c] = s2[0]; }
}

__global__ void bn_fin_kernel(const float* __restrict__ sum, const float* __restrict__ sq,
                              const float* __restrict__ g, const float* __restrict__ be,
                              float* __restrict__ scale, float* __restrict__ shift,
                              int nch, float inv_cnt) {
  int c = blockIdx.x * blockDim.x + threadIdx.x;
  if (c >= nch) return;
  float m = sum[c] * inv_cnt;
  float v = sq[c] * inv_cnt - m * m;
  float sc = g[c] * rsqrtf(v + EPS_);
  scale[c] = sc;
  shift[c] = be[c] - m * sc;
}

__global__ void bn_pack_pos_kernel(const float* __restrict__ p1t, const float* __restrict__ scale,
                                   const float* __restrict__ shift, bf16* __restrict__ p1b) {
  int i = blockIdx.x * blockDim.x + threadIdx.x;   // B*NK*64
  int c = i & 63;
  p1b[i] = (bf16)fmaxf(p1t[i] * scale[c] + shift[c], 0.f);
}

// pos_emb = pos_w2 @ relu(bn(p1)) + b2 -> peT f32 (B*NK, 256)
__global__ __launch_bounds__(256)
void gemm_pe_kernel(const bf16* __restrict__ wbf, const bf16* __restrict__ p1b,
                    const float* __restrict__ pb2, float* __restrict__ peT) {
  int gw = (blockIdx.x * blockDim.x + threadIdx.x) >> 5;
  int b = gw >> 16, t = gw & 65535;                // 16*4096 tiles per batch
  int m0 = (t & 15) << 4, c0 = (t >> 4) << 4;
  const bf16* X = p1b + (size_t)b * NK_ * PH_;
  v8f acc = {};
  for (int k = 0; k < 64; k += 32)
    acc = wmma_bf16(load_frag(wbf + OW_POSW2, 64, m0, k), load_frag(X, 64, c0, k), acc);
  int lane = threadIdx.x & 31, r = lane & 15, hi = lane >> 4;
  float* out = peT + ((size_t)b * NK_ + (c0 + r)) * DIM_;
#pragma unroll
  for (int i = 0; i < 8; ++i) {
    int row = m0 + i + 8 * hi;
    out[row] = acc[i] + pb2[row];
  }
}

// sT[b][col][c] = q_f[c,n] - k_f[c,idx] + pos_emb[c,col]   (bf16, attn-MLP input)
__global__ void s_pack_kernel(const float* __restrict__ qT, const float* __restrict__ kT,
                              const float* __restrict__ peT, const int* __restrict__ idx,
                              bf16* __restrict__ sT) {
  int i = blockIdx.x * blockDim.x + threadIdx.x;   // B*NK*256
  int c = i & 255;
  int col = (i >> 8) & (NK_ - 1);
  int b = i >> 24;
  int n = col >> 4;
  int j = idx[(size_t)b * NK_ + col];
  float v = qT[((size_t)b * N_ + n) * DIM_ + c] - kT[((size_t)b * N_ + j) * DIM_ + c] +
            peT[(size_t)i];
  sT[i] = (bf16)v;
}

// ---------------- attn MLP pass 1: BN statistics of h = att_w1 @ s + b1 ----------------
// One block per 16-column tile. The sT column panel is async-staged to LDS once
// and the 8 B fragments are register-cached across the 8 row tiles of each wave.
// Each wave owns 128 disjoint rows -> race-free, deterministic partials.
__global__ __launch_bounds__(256)
void att_pass1_kernel(const bf16* __restrict__ wbf, const bf16* __restrict__ sT,
                      const float* __restrict__ ab1, float* __restrict__ part) {
  __shared__ bf16 sS[16 * DIM_];                   // 8 KB staged panel
  int b = blockIdx.x >> 12;                        // 4096 col tiles per batch
  int c0 = (blockIdx.x & 4095) << 4;
  const bf16* X = sT + ((size_t)b * NK_ + c0) * DIM_;
  async_stage(X, sS, 16 * DIM_ * 2);
  __syncthreads();
  int wave = threadIdx.x >> 5, lane = threadIdx.x & 31, r = lane & 15, hi = lane >> 4;
  const bf16* W = wbf + OW_ATTW1;
  float* pblk = part + (size_t)blockIdx.x * 2048;
  v16bf bfr[8];
#pragma unroll
  for (int k = 0; k < 8; ++k) bfr[k] = load_frag(sS, DIM_, 0, k * 32);
  for (int rt = 0; rt < 8; ++rt) {
    int m0 = (wave * 8 + rt) << 4;
    v8f acc = {};
#pragma unroll
    for (int k = 0; k < 8; ++k)
      acc = wmma_bf16(load_frag(W, DIM_, m0, k * 32), bfr[k], acc);
#pragma unroll
    for (int i = 0; i < 8; ++i) {
      int row = m0 + i + 8 * hi;
      float h = acc[i] + ab1[row];
      float s1 = red16_sum(h);
      float s2 = red16_sum(h * h);
      if (r == 0) { pblk[row] = s1; pblk[1024 + row] = s2; }
    }
  }
}

// fixed-order reduce of the per-block partials (deterministic)
__global__ void att_stats_reduce_kernel(const float* __restrict__ part, float* __restrict__ sum,
                                        float* __restrict__ sq) {
  int ch = blockIdx.x * blockDim.x + threadIdx.x;
  if (ch >= AH_) return;
  float a = 0.f, q = 0.f;
  for (int bk = 0; bk < B_ * 4096; ++bk) {
    a += part[(size_t)bk * 2048 + ch];
    q += part[(size_t)bk * 2048 + 1024 + ch];
  }
  sum[ch] = a; sq[ch] = q;
}

// ---------------- attn MLP pass 2, fused ----------------
// async-stage sT panel (32 KB) -> GEMM1 (A frags register-cached per row tile,
// B from LDS) -> BN/ReLU -> hT in LDS (128 KB) -> GEMM2 (A frag reused across 4
// column accumulators, B from LDS) -> softmax over the 16 neighbors (shfl) ->
// agg = sum_k attn*(v_f+pos_emb) -> aggT bf16.  LDS total 160 KB (CDNA5 WGP).
__global__ __launch_bounds__(256)
void att_pass2_kernel(const bf16* __restrict__ wbf, const bf16* __restrict__ sT,
                      const float* __restrict__ ab1, const float* __restrict__ ab2,
                      const float* __restrict__ ascale, const float* __restrict__ ashift,
                      const float* __restrict__ vT, const float* __restrict__ peT,
                      bf16* __restrict__ aggT) {
  extern __shared__ bf16 smem[];
  bf16* hT = smem;                                 // [64 local cols][1024]
  bf16* sS = smem + 64 * AH_;                      // [64 local cols][256]
  int b = blockIdx.x >> 10;                        // 1024 blocks per batch (4 points each)
  int colbase = (blockIdx.x & 1023) * 64;          // global column base (n*16+k flat)
  const bf16* X = sT + ((size_t)b * NK_ + colbase) * DIM_;
  async_stage(X, sS, 64 * DIM_ * 2);
  __syncthreads();
  int wave = threadIdx.x >> 5, lane = threadIdx.x & 31, r = lane & 15, hi = lane >> 4;
  const bf16* W1 = wbf + OW_ATTW1;
  // stage 1: each wave covers 8 row tiles x 4 col tiles; A frags cached per row tile
  for (int rt = 0; rt < 8; ++rt) {
    int m0 = (wave * 8 + rt) << 4;
    v16bf afr[8];
#pragma unroll
    for (int k = 0; k < 8; ++k) afr[k] = load_frag(W1, DIM_, m0, k * 32);
#pragma unroll
    for (int ct = 0; ct < 4; ++ct) {
      v8f acc = {};
#pragma unroll
      for (int k = 0; k < 8; ++k)
        acc = wmma_bf16(afr[k], load_frag(sS, DIM_, ct << 4, k * 32), acc);
      int lc = (ct << 4) + r;
#pragma unroll
      for (int i = 0; i < 8; ++i) {
        int row = m0 + i + 8 * hi;
        float h = acc[i] + ab1[row];
        h = fmaxf(h * ascale[row] + ashift[row], 0.f);
        hT[(size_t)lc * AH_ + row] = (bf16)h;
      }
    }
  }
  __syncthreads();
  // stage 2: 16 row tiles over 8 waves; one A frag feeds 4 column accumulators
  const bf16* W2 = wbf + OW_ATTW2;
  for (int rr = 0; rr < 2; ++rr) {
    int m0 = (wave * 2 + rr) << 4;
    v8f acc[4] = {{}, {}, {}, {}};
    for (int k = 0; k < AH_; k += 32) {
      v16bf a = load_frag(W2, AH_, m0, k);
#pragma unroll
      for (int ct = 0; ct < 4; ++ct)
        acc[ct] = wmma_bf16(a, load_frag(hT, AH_, ct << 4, k), acc[ct]);
    }
#pragma unroll
    for (int ct = 0; ct < 4; ++ct) {
      int gcol = colbase + (ct << 4) + r;          // each 16-col tile = one point's K
      int n = gcol >> 4;
      const float* vrow = vT + ((size_t)b * N_ + n) * DIM_;
      const float* perow = peT + ((size_t)b * NK_ + gcol) * DIM_;
#pragma unroll
      for (int i = 0; i < 8; ++i) {
        int row = m0 + i + 8 * hi;
        float x = acc[ct][i] + ab2[row];
        float m = red16_max(x);
        float e = __expf(x - m);
        float s = red16_sum(e);
        float a2 = e / s;                          // softmax over the 16 neighbors
        float contrib = red16_sum(a2 * (vrow[row] + perow[row]));
        if (r == 0) aggT[((size_t)b * N_ + n) * DIM_ + row] = (bf16)contrib;
      }
    }
  }
}

// ---------------- final: y = we @ agg + be + identity(value) ----------------
__global__ __launch_bounds__(256)
void gemm_final_kernel(const bf16* __restrict__ wbf, const bf16* __restrict__ aggT,
                       const float* __restrict__ be, const float* __restrict__ value,
                       float* __restrict__ out) {
  int gw = (blockIdx.x * blockDim.x + threadIdx.x) >> 5;
  int b = gw >> 11, t = gw & 2047;
  int m0 = (t & 7) << 4, c0 = (t >> 3) << 4;
  const bf16* X = aggT + (size_t)b * N_ * DIM_;
  v8f acc = {};
  for (int k = 0; k < 256; k += 32)
    acc = wmma_bf16(load_frag(wbf + OW_WE, 256, m0, k), load_frag(X, 256, c0, k), acc);
  int lane = threadIdx.x & 31, r = lane & 15, hi = lane >> 4;
  int col = c0 + r;
#pragma unroll
  for (int i = 0; i < 8; ++i) {
    int row = m0 + i + 8 * hi;
    size_t o = ((size_t)b * C_ + row) * N_ + col;
    out[o] = acc[i] + be[row] + value[o];
  }
}

// ---------------- host ----------------
extern "C" void kernel_launch(void* const* d_in, const int* in_sizes, int n_in,
                              void* d_out, int out_size, void* d_ws, size_t ws_size,
                              hipStream_t stream) {
  (void)in_sizes; (void)n_in; (void)out_size; (void)ws_size;
  const float* pos     = (const float*)d_in[0];
  const float* key     = (const float*)d_in[1];
  const float* query   = (const float*)d_in[2];
  const float* mlpv_w1 = (const float*)d_in[3];  const float* mlpv_b1 = (const float*)d_in[4];
  const float* mlpv_w2 = (const float*)d_in[5];  const float* mlpv_b2 = (const float*)d_in[6];
  const float* mlpv_ws = (const float*)d_in[7];  const float* mlpv_bs = (const float*)d_in[8];
  const float* wk = (const float*)d_in[9];   const float* bk = (const float*)d_in[10];
  const float* wq = (const float*)d_in[11];  const float* bq = (const float*)d_in[12];
  const float* wv = (const float*)d_in[13];  const float* bv = (const float*)d_in[14];
  const float* pos_w1 = (const float*)d_in[15]; const float* pos_b1 = (const float*)d_in[16];
  const float* pos_g1 = (const float*)d_in[17]; const float* pos_be1 = (const float*)d_in[18];
  const float* pos_w2 = (const float*)d_in[19]; const float* pos_b2 = (const float*)d_in[20];
  const float* att_w1 = (const float*)d_in[21]; const float* att_b1 = (const float*)d_in[22];
  const float* att_g1 = (const float*)d_in[23]; const float* att_be1 = (const float*)d_in[24];
  const float* att_w2 = (const float*)d_in[25]; const float* att_b2 = (const float*)d_in[26];
  const float* we = (const float*)d_in[27];  const float* be = (const float*)d_in[28];
  float* out = (float*)d_out;

  char* ws = (char*)d_ws;
  size_t off = 0;
  auto take = [&](size_t bytes) -> char* {
    char* p = ws + off;
    off = (off + bytes + 255) & ~(size_t)255;
    return p;
  };
  bf16*  wbf   = (bf16*) take((size_t)W_TOTAL * 2);
  bf16*  xT    = (bf16*) take((size_t)B_ * N_ * 256 * 2);
  bf16*  h1T   = (bf16*) take((size_t)B_ * N_ * C_ * 2);
  float* value = (float*)take((size_t)B_ * C_ * N_ * 4);
  bf16*  valT  = (bf16*) take((size_t)B_ * N_ * C_ * 2);
  float* qT    = (float*)take((size_t)B_ * N_ * DIM_ * 4);
  float* kT    = (float*)take((size_t)B_ * N_ * DIM_ * 4);
  float* vT    = (float*)take((size_t)B_ * N_ * DIM_ * 4);
  int*   idx   = (int*)  take((size_t)B_ * N_ * K_ * 4);
  float* p1t   = (float*)take((size_t)B_ * NK_ * PH_ * 4);
  bf16*  p1b   = (bf16*) take((size_t)B_ * NK_ * PH_ * 2);
  float* peT   = (float*)take((size_t)B_ * NK_ * DIM_ * 4);
  bf16*  sT    = (bf16*) take((size_t)B_ * NK_ * DIM_ * 2);
  bf16*  aggT  = (bf16*) take((size_t)B_ * N_ * DIM_ * 2);
  float* apart = (float*)take((size_t)B_ * 4096 * 2048 * 4);   // pass-1 partials
  float* stats = (float*)take((size_t)4352 * 4);
  float* pos_sum = stats,        * pos_sq  = stats + 64;
  float* pos_scl = stats + 128,  * pos_sh  = stats + 192;
  float* att_sum = stats + 256,  * att_sq  = stats + 1280;
  float* att_scl = stats + 2304, * att_sh  = stats + 3328;

  const int TB = 256;
#define CVT(src, o, n) cvt_f32_bf16_kernel<<<((n) + TB - 1) / TB, TB, 0, stream>>>(src, wbf + (o), (n))
  CVT(mlpv_w1, OW_MLPV_W1, 32768);  CVT(mlpv_w2, OW_MLPV_W2, 16384);
  CVT(mlpv_ws, OW_MLPV_WS, 32768);  CVT(wk, OW_WK, 32768);
  CVT(wq, OW_WQ, 32768);            CVT(wv, OW_WV, 32768);
  CVT(pos_w2, OW_POSW2, 16384);     CVT(att_w1, OW_ATTW1, 262144);
  CVT(att_w2, OW_ATTW2, 262144);    CVT(we, OW_WE, 32768);
#undef CVT

  pack_x_kernel<<<(B_ * N_ * 256) / TB, TB, 0, stream>>>(key, query, xT);
  gemm_h1_kernel<<<(B_ * 2048) / 8, TB, 0, stream>>>(wbf, xT, mlpv_b1, h1T);
  gemm_value_kernel<<<(B_ * 2048) / 8, TB, 0, stream>>>(wbf, xT, h1T, mlpv_b2, mlpv_bs,
                                                        value, valT);
  gemm_proj_kernel<<<(B_ * 4096) / 8, TB, 0, stream>>>(wbf + OW_WK, xT, 256, bk, kT);
  gemm_proj_kernel<<<(B_ * 4096) / 8, TB, 0, stream>>>(wbf + OW_WQ, xT + 128, 256, bq, qT);
  gemm_proj_kernel<<<(B_ * 4096) / 8, TB, 0, stream>>>(wbf + OW_WV, valT, 128, bv, vT);
  knn_kernel<<<B_ * (N_ / 256), TB, 0, stream>>>(pos, idx);
  pos_hidden_kernel<<<(B_ * NK_) / TB, TB, 0, stream>>>(pos, idx, pos_w1, pos_b1, p1t);
  pos_stats_kernel<<<PH_, TB, 0, stream>>>(p1t, pos_sum, pos_sq);
  bn_fin_kernel<<<1, 64, 0, stream>>>(pos_sum, pos_sq, pos_g1, pos_be1, pos_scl, pos_sh,
                                      PH_, 1.f / (float)(B_ * NK_));
  bn_pack_pos_kernel<<<(B_ * NK_ * PH_) / TB, TB, 0, stream>>>(p1t, pos_scl, pos_sh, p1b);
  gemm_pe_kernel<<<(B_ * 65536) / 8, TB, 0, stream>>>(wbf, p1b, pos_b2, peT);
  s_pack_kernel<<<(B_ * NK_ * DIM_) / TB, TB, 0, stream>>>(qT, kT, peT, idx, sT);
  att_pass1_kernel<<<B_ * 4096, TB, 0, stream>>>(wbf, sT, att_b1, apart);
  att_stats_reduce_kernel<<<AH_ / TB, TB, 0, stream>>>(apart, att_sum, att_sq);
  bn_fin_kernel<<<4, TB, 0, stream>>>(att_sum, att_sq, att_g1, att_be1, att_scl, att_sh,
                                      AH_, 1.f / (float)(B_ * NK_));
  att_pass2_kernel<<<B_ * (N_ / 4), TB, (64 * AH_ + 64 * DIM_) * sizeof(bf16), stream>>>(
      wbf, sT, att_b1, att_b2, att_scl, att_sh, vT, peT, aggT);
  gemm_final_kernel<<<(B_ * 2048) / 8, TB, 0, stream>>>(wbf, aggT, be, value, out);
}
